// MambaBlock_4243427689265
// MI455X (gfx1250) — matrix operand order repaired and checked by reading, not compiled, not run on hardware
//
#include <hip/hip_runtime.h>
#include <hip/hip_bf16.h>
#include <stdint.h>

// ---------------------------------------------------------------------------
// MambaBlock forward for MI455X (gfx1250, wave32, WMMA).
// GEMMs: bf16 WMMA (v_wmma_f32_16x16x32_bf16, f32 accum) with LDS-tiled,
// double-buffered mainloop staged by async global->LDS copies (ASYNCcnt).
// Elementwise conv/SiLU/softplus and the L-sequential SSM scan are fused.
// ---------------------------------------------------------------------------

typedef __attribute__((ext_vector_type(16))) __bf16 v16bf;
typedef __attribute__((ext_vector_type(8)))  __bf16 v8bf;
typedef __attribute__((ext_vector_type(8)))  float  v8f;
typedef __attribute__((ext_vector_type(4)))  float  v4f;
typedef int v4i32 __attribute__((__vector_size__(4 * sizeof(int))));  // builtin's int4

#define AS1 __attribute__((address_space(1)))
#define AS3 __attribute__((address_space(3)))

#if defined(__HIP_DEVICE_COMPILE__) && __has_builtin(__builtin_amdgcn_global_load_async_to_lds_b128)
#define HAVE_ASYNC_LDS 1
#else
#define HAVE_ASYNC_LDS 0
#endif

static __device__ __forceinline__ __bf16 f32_to_bf16(float f) {
    // round-to-nearest-even fp32 -> bf16
    uint32_t u = __builtin_bit_cast(uint32_t, f);
    uint32_t r = u + 0x7FFFu + ((u >> 16) & 1u);
    unsigned short hi = (unsigned short)(r >> 16);
    return __builtin_bit_cast(__bf16, hi);
}

static __device__ __forceinline__ float silu_f(float x) {
    return x / (1.0f + __expf(-x));
}

// Copy one 16-element (32B) bf16 chunk global -> LDS.
// Async path: global_load_async_to_lds_b128 x2 (per-lane, tracked by ASYNCcnt).
// Builtin prototype (from hipcc): (int4 AS1*, int4 AS3*, imm offset, imm cpol).
// LDS raw offset = low 32 bits of the generic pointer (flat-LDS aperture keeps
// the wave-relative offset in addr[31:0]); global AS1 VA == flat VA.
static __device__ __forceinline__ void stage_chunk(const __bf16* src, __bf16* dst) {
#if HAVE_ASYNC_LDS
    AS1 v4i32* g0 = (AS1 v4i32*)(size_t)(const void*)src;
    AS1 v4i32* g1 = (AS1 v4i32*)(size_t)(const void*)(src + 8);
    AS3 v4i32* l0 = (AS3 v4i32*)(uint32_t)(size_t)(void*)dst;
    AS3 v4i32* l1 = (AS3 v4i32*)(uint32_t)(size_t)(void*)(dst + 8);
    __builtin_amdgcn_global_load_async_to_lds_b128(g0, l0, 0, 0);
    __builtin_amdgcn_global_load_async_to_lds_b128(g1, l1, 0, 0);
#else
    *(v8bf*)dst       = *(const v8bf*)src;
    *(v8bf*)(dst + 8) = *(const v8bf*)(src + 8);
#endif
}

static __device__ __forceinline__ void async_wait() {
#if HAVE_ASYNC_LDS
#if __has_builtin(__builtin_amdgcn_s_wait_asynccnt)
    __builtin_amdgcn_s_wait_asynccnt(0);
#else
    asm volatile("s_wait_asynccnt 0x0" ::: "memory");
#endif
#endif
}

// ---------------------------------------------------------------------------
// fp32 -> bf16 cast (vectorized b128 loads)
// ---------------------------------------------------------------------------
__global__ void cast_f32_bf16_kernel(const float* __restrict__ in,
                                     __bf16* __restrict__ out, int n) {
    int i = (blockIdx.x * blockDim.x + threadIdx.x) * 4;
    if (i + 3 < n) {
        v4f v = *(const v4f*)(in + i);
        out[i + 0] = f32_to_bf16(v.x);
        out[i + 1] = f32_to_bf16(v.y);
        out[i + 2] = f32_to_bf16(v.z);
        out[i + 3] = f32_to_bf16(v.w);
    } else {
        for (; i < n; ++i) out[i] = f32_to_bf16(in[i]);
    }
}

// ---------------------------------------------------------------------------
// LDS-tiled bf16 WMMA GEMM: D[M,N] = epilogue( A[M,K] @ W[N,K]^T )
//   Block: 256 threads = 8 waves as 2(M) x 4(N). Block tile 128x128, K-step 32.
//   Wave tile 64x32: 4(M) x 2(N) accumulators -> 8 v_wmma per k-step.
//   LDS: A/B tiles 128x32 bf16, rows padded to 40 elems (80B) so row-strided
//   16-lane ds_load_b128 fragment reads are bank-conflict free. Double buffered
//   (40KB), next tile staged via async copies while current tile is multiplied.
// Fragment layouts used here (wave32, same contiguous k order for A and B):
//   A lane L (g=L/16): row = L%16, halves 0..15 = K[16g..16g+15] (contiguous)
//   B lane L: col = L%16, halves 0..15 = K[16g..16g+15] (contiguous)
//   C/D lane L, vgpr r: element (r + 8g, L%16)
// ---------------------------------------------------------------------------
enum {
    EP_BIAS     = 1,   // + bias[n]
    EP_ACCUM    = 2,   // += existing outF
    EP_SILUGATE = 4,   // out = aux * silu(acc)      (gated residual z)
    EP_RESID    = 8,   // out = aux + acc            (final residual add)
    EP_OUTBF16  = 16,  // write bf16 to outB instead of f32 to outF
};

#define BM 128
#define BN 128
#define BK 32
#define LDSS 40   // padded LDS row stride (elements)

template <int EP>
__global__ __launch_bounds__(256)
void wmma_gemm_bf16(const __bf16* __restrict__ A, const __bf16* __restrict__ W,
                    const float* __restrict__ bias, const float* __restrict__ aux,
                    float* __restrict__ outF, __bf16* __restrict__ outB,
                    int M, int N, int K) {
    __shared__ __bf16 sA[2][BM * LDSS];
    __shared__ __bf16 sB[2][BN * LDSS];

    const int tid  = threadIdx.x;
    const int lane = tid & 31;
    const int wid  = tid >> 5;
    const int g    = lane >> 4;       // half-wave group (0/1)
    const int ln   = lane & 15;
    const int wm   = (wid & 1) * 64;  // wave M offset in block tile
    const int wn   = (wid >> 1) * 32; // wave N offset in block tile
    const int m0   = blockIdx.x * BM;
    const int n0   = blockIdx.y * BN;

    // global->LDS staging: 256 threads x 32B cover one 128x32 bf16 tile
    const int cr = tid >> 1;            // copy row (0..127)
    const int ch = (tid & 1) * 16;      // copy K-half (0 or 16 elements)
    const __bf16* gA = A + (size_t)(m0 + cr) * K + ch;
    const __bf16* gW = W + (size_t)(n0 + cr) * K + ch;
    __bf16* dA = &sA[0][0] + cr * LDSS + ch;
    __bf16* dB = &sB[0][0] + cr * LDSS + ch;
    const int bufStride = BM * LDSS;

    v8f acc[4][2] = {};

    // prologue: stage k=0 tile into buffer 0
    stage_chunk(gA, dA);
    stage_chunk(gW, dB);
    async_wait();
    __syncthreads();

    int cur = 0;
    for (int kk = 0; kk < K; kk += BK) {
        if (kk + BK < K) {  // stage next K-tile into the other buffer
            int nxt = cur ^ 1;
            stage_chunk(gA + kk + BK, dA + nxt * bufStride);
            stage_chunk(gW + kk + BK, dB + nxt * bufStride);
            if (kk + 2 * BK < K) {
                __builtin_prefetch(gA + kk + 2 * BK, 0, 0);  // global_prefetch_b8
                __builtin_prefetch(gW + kk + 2 * BK, 0, 0);
            }
        }
        const __bf16* pa = &sA[cur][0];
        const __bf16* pb = &sB[cur][0];

        union { v16bf v; v8bf h[2]; } bfr[2];
        #pragma unroll
        for (int j = 0; j < 2; ++j) {
            const __bf16* p = pb + (wn + 16 * j + ln) * LDSS + 16 * g;
            bfr[j].h[0] = *(const v8bf*)p;
            bfr[j].h[1] = *(const v8bf*)(p + 8);
        }
        #pragma unroll
        for (int i = 0; i < 4; ++i) {
            union { v16bf v; v8bf h[2]; } af;
            const __bf16* p = pa + (wm + 16 * i + ln) * LDSS + 16 * g;
            af.h[0] = *(const v8bf*)p;
            af.h[1] = *(const v8bf*)(p + 8);
            acc[i][0] = __builtin_amdgcn_wmma_f32_16x16x32_bf16(
                false, af.v, false, bfr[0].v, (short)0, acc[i][0], false, false);
            acc[i][1] = __builtin_amdgcn_wmma_f32_16x16x32_bf16(
                false, af.v, false, bfr[1].v, (short)0, acc[i][1], false, false);
        }
        async_wait();
        __syncthreads();
        cur ^= 1;
    }

    float bv[2] = {0.0f, 0.0f};
    if (EP & EP_BIAS) {
        bv[0] = bias[n0 + wn + ln];
        bv[1] = bias[n0 + wn + 16 + ln];
    }
    #pragma unroll
    for (int i = 0; i < 4; ++i) {
        #pragma unroll
        for (int j = 0; j < 2; ++j) {
            #pragma unroll
            for (int r = 0; r < 8; ++r) {
                int m = m0 + wm + 16 * i + 8 * g + r;
                int n = n0 + wn + 16 * j + ln;
                size_t idx = (size_t)m * N + n;
                float v = acc[i][j][r] + bv[j];
                if (EP & EP_ACCUM)    v += outF[idx];
                if (EP & EP_SILUGATE) v  = aux[idx] * silu_f(v);
                if (EP & EP_RESID)    v += aux[idx];
                if (EP & EP_OUTBF16)  outB[idx] = f32_to_bf16(v);
                else                  outF[idx] = v;
            }
        }
    }
}

// ---------------------------------------------------------------------------
// Depthwise conv1d (k=3, pad=1, per-channel) + SiLU; writes fp32 + bf16 copies.
// ---------------------------------------------------------------------------
__global__ void conv_silu_kernel(const float* __restrict__ x1,
                                 const float* __restrict__ cw,
                                 const float* __restrict__ cb,
                                 float* __restrict__ xa,
                                 __bf16* __restrict__ xab,
                                 int L, int d, long total) {
    long i = (long)blockIdx.x * blockDim.x + threadIdx.x;
    if (i >= total) return;
    int c = (int)(i % d);
    long bl = i / d;
    int l = (int)(bl % L);
    float w0 = cw[c * 3 + 0], w1 = cw[c * 3 + 1], w2 = cw[c * 3 + 2];
    float acc = cb[c];
    float xm = (l > 0)     ? x1[i - d] : 0.0f;   // cross-correlation: tap0 -> l-1
    float xc = x1[i];
    float xp = (l < L - 1) ? x1[i + d] : 0.0f;
    acc += w0 * xm + w1 * xc + w2 * xp;
    float s = silu_f(acc);
    xa[i]  = s;
    xab[i] = f32_to_bf16(s);
}

// ---------------------------------------------------------------------------
// SSM selective-scan: one thread per (batch, hidden-channel); recurrence
//   delta = softplus(dly); Abar = exp(delta*A[h]); h = Abar*h + delta*xb
// (identical to the reference cumprod/cumsum form). Loads are batched 4 steps
// ahead of the serial recurrence to hide latency at low occupancy.
// ---------------------------------------------------------------------------
__global__ void ssm_scan_kernel(const float* __restrict__ dly,
                                const float* __restrict__ xb,
                                const float* __restrict__ Avec,
                                __bf16* __restrict__ hbf,
                                int Bn, int L, int H) {
    int t = blockIdx.x * blockDim.x + threadIdx.x;
    if (t >= Bn * H) return;
    int b = t / H, hc = t % H;
    float Ah = Avec[hc];
    float h  = 0.0f;
    size_t base = (size_t)b * L * H + hc;
    for (int l = 0; l < L; l += 4) {
        float dl[4], xv[4];
        #pragma unroll
        for (int u = 0; u < 4; ++u) {         // issue all 8 loads as one clause
            size_t idx = base + (size_t)(l + u) * H;
            dl[u] = dly[idx];
            xv[u] = xb[idx];
        }
        #pragma unroll
        for (int u = 0; u < 4; ++u) {
            float d0 = dl[u];
            d0 = (d0 > 20.0f) ? d0 : __logf(1.0f + __expf(d0));  // softplus
            float ab  = __expf(d0 * Ah);
            h = ab * h + d0 * xv[u];
            hbf[base + (size_t)(l + u) * H] = f32_to_bf16(h);
        }
    }
}

// ---------------------------------------------------------------------------
// Host-side orchestration
// ---------------------------------------------------------------------------
static inline size_t align256(size_t x) { return (x + 255) & ~(size_t)255; }

extern "C" void kernel_launch(void* const* d_in, const int* in_sizes, int n_in,
                              void* d_out, int out_size, void* d_ws, size_t ws_size,
                              hipStream_t stream) {
    (void)in_sizes; (void)n_in; (void)out_size;
    const int Bn = 2, L = 4096, d = 1024, h = 2048;
    const long Nrow = (long)Bn * L;               // 8192 rows

    const float* x      = (const float*)d_in[0];
    const float* W1     = (const float*)d_in[1];
    const float* conv_w = (const float*)d_in[2];
    const float* conv_b = (const float*)d_in[3];
    const float* Avec   = (const float*)d_in[4];
    const float* Wb     = (const float*)d_in[5];
    const float* bb     = (const float*)d_in[6];
    const float* Wc     = (const float*)d_in[7];
    const float* bc     = (const float*)d_in[8];
    const float* Wd     = (const float*)d_in[9];
    const float* bd     = (const float*)d_in[10];
    const float* Wdelta = (const float*)d_in[11];
    const float* bdelta = (const float*)d_in[12];
    const float* W2     = (const float*)d_in[13];
    const float* Wlast  = (const float*)d_in[14];
    float* out = (float*)d_out;

    // ---- workspace arena ----
    char* ws = (char*)d_ws;
    size_t off = 0;
    auto alloc = [&](size_t bytes) -> void* {
        void* p = ws + off;
        off = align256(off + bytes);
        return p;
    };
    __bf16* xbf  = (__bf16*)alloc((size_t)Nrow * d * 2);
    __bf16* w1b  = (__bf16*)alloc((size_t)d * d * 2);
    __bf16* wdlb = (__bf16*)alloc((size_t)h * d * 2);
    __bf16* wbb  = (__bf16*)alloc((size_t)h * d * 2);
    __bf16* wcb  = (__bf16*)alloc((size_t)d * h * 2);
    __bf16* wddb = (__bf16*)alloc((size_t)d * d * 2);
    __bf16* w2b  = (__bf16*)alloc((size_t)d * d * 2);
    __bf16* wlb  = (__bf16*)alloc((size_t)d * d * 2);
    float*  x1   = (float*)alloc((size_t)Nrow * d * 4);   // reused as oss later
    float*  xa   = (float*)alloc((size_t)Nrow * d * 4);
    __bf16* xab  = (__bf16*)alloc((size_t)Nrow * d * 2);
    float*  dly  = (float*)alloc((size_t)Nrow * h * 4);
    float*  xb   = (float*)alloc((size_t)Nrow * h * 4);
    __bf16* hbf  = (__bf16*)alloc((size_t)Nrow * h * 2);
    __bf16* osb  = (__bf16*)alloc((size_t)Nrow * d * 2);
    __bf16* zb   = (__bf16*)alloc((size_t)Nrow * d * 2);
    float*  oss  = x1;                                    // alias: x1 dead post-conv
    if (off > ws_size) return;

    auto cast = [&](const float* src, __bf16* dst, long n) {
        int blocks = (int)((n / 4 + 255) / 256);
        hipLaunchKernelGGL(cast_f32_bf16_kernel, dim3(blocks), dim3(256), 0, stream,
                           src, dst, (int)n);
    };

    cast(x, xbf, Nrow * d);
    cast(W1, w1b, (long)d * d);
    cast(Wdelta, wdlb, (long)h * d);
    cast(Wb, wbb, (long)h * d);
    cast(Wc, wcb, (long)d * h);
    cast(Wd, wddb, (long)d * d);
    cast(W2, w2b, (long)d * d);
    cast(Wlast, wlb, (long)d * d);

    const dim3 blk(256);
    const dim3 gd_d(Nrow / BM, d / BN);   // 64 x 8
    const dim3 gd_h(Nrow / BM, h / BN);   // 64 x 16

    // G1: x1 = x @ W1^T
    hipLaunchKernelGGL((wmma_gemm_bf16<0>), gd_d, blk, 0, stream,
                       xbf, w1b, nullptr, nullptr, x1, nullptr, (int)Nrow, d, d);

    // depthwise conv + SiLU -> xa (f32 + bf16)
    {
        long total = Nrow * d;
        int blocks = (int)((total + 255) / 256);
        hipLaunchKernelGGL(conv_silu_kernel, dim3(blocks), blk, 0, stream,
                           x1, conv_w, conv_b, xa, xab, L, d, total);
    }

    // G2: dly = xa @ Wdelta^T + bdelta (pre-softplus)
    hipLaunchKernelGGL((wmma_gemm_bf16<EP_BIAS>), gd_h, blk, 0, stream,
                       xab, wdlb, bdelta, nullptr, dly, nullptr, (int)Nrow, h, d);
    // G3: xb = xa @ Wb^T + bb
    hipLaunchKernelGGL((wmma_gemm_bf16<EP_BIAS>), gd_h, blk, 0, stream,
                       xab, wbb, bb, nullptr, xb, nullptr, (int)Nrow, h, d);

    // SSM scan -> hbf (bf16)
    hipLaunchKernelGGL(ssm_scan_kernel, dim3((Bn * h + 255) / 256), blk, 0, stream,
                       dly, xb, Avec, hbf, Bn, L, h);

    // G4: oss = h_st @ Wc^T + bc
    hipLaunchKernelGGL((wmma_gemm_bf16<EP_BIAS>), gd_d, blk, 0, stream,
                       hbf, wcb, bc, nullptr, oss, nullptr, (int)Nrow, d, h);
    // G5: oss += xa @ Wd^T + bd
    hipLaunchKernelGGL((wmma_gemm_bf16<EP_BIAS | EP_ACCUM>), gd_d, blk, 0, stream,
                       xab, wddb, bd, nullptr, oss, nullptr, (int)Nrow, d, d);

    // oss -> bf16
    cast(oss, osb, Nrow * d);

    // G6: z = oss * silu(oss @ W2^T)  (bf16 out)
    hipLaunchKernelGGL((wmma_gemm_bf16<EP_SILUGATE | EP_OUTBF16>), gd_d, blk, 0, stream,
                       osb, w2b, nullptr, oss, nullptr, zb, (int)Nrow, d, d);

    // G7: out = oss + z @ Wlast^T
    hipLaunchKernelGGL((wmma_gemm_bf16<EP_RESID>), gd_d, blk, 0, stream,
                       zb, wlb, nullptr, oss, out, nullptr, (int)Nrow, d, d);
}
